// EnhancedMultiHeadAttention_25666724561083
// MI455X (gfx1250) — matrix operand order repaired and checked by reading, not compiled, not run on hardware
//
#include <hip/hip_runtime.h>
#include <hip/hip_bf16.h>

// EnhancedMultiHeadAttention for MI455X (gfx1250, wave32, WMMA f16->f32).
// B=2, S=2048, D=1024, H=16, HD=64.
// All four matmul stages run on v_wmma_f32_16x16x32_f16 with f32 accumulate;
// K-loops are software-pipelined (double-buffered fragments) so VMEM overlaps
// the WMMA issue instead of s_wait_loadcnt 0 before every matrix op.

typedef __attribute__((ext_vector_type(16))) _Float16 v16h;
typedef __attribute__((ext_vector_type(8)))  _Float16 v8h;
typedef __attribute__((ext_vector_type(8)))  float    v8f;

#define B_  2
#define S_  2048
#define D_  1024
#define H_  16
#define HD_ 64

static __device__ inline v8f wmma16x16x32(v16h a, v16h b, v8f c) {
  return __builtin_amdgcn_wmma_f32_16x16x32_f16(false, a, false, b, (short)0, c,
                                                false, false);
}

// A-matrix fragment (16x32, MxK) from a row-major f16 matrix, row stride `ld`
// (halves). Lane L (row=L%16, grp=L/16): VGPR0-3 = K grp*8..+7,
// VGPR4-7 = K 16+grp*8..+7. Two 16-byte loads per lane.
static __device__ inline v16h load_A_frag(const _Float16* base, int ld) {
  int lane = threadIdx.x & 31;
  const _Float16* p = base + (size_t)(lane & 15) * ld + (lane >> 4) * 8;
  v8h lo = *(const v8h*)p;
  v8h hi = *(const v8h*)(p + 16);
  v16h r;
#pragma unroll
  for (int i = 0; i < 8; ++i) { r[i] = lo[i]; r[8 + i] = hi[i]; }
  return r;
}

// B-matrix fragment (32x16, KxN) from a TRANSPOSED row-major matrix BT[n][k],
// row stride `ld` (halves). Lane L (n=L%16): K grp*8..+7, then K 16+grp*8..+7.
static __device__ inline v16h load_BT_frag(const _Float16* baseT, int ld) {
  int lane = threadIdx.x & 31;
  const _Float16* p = baseT + (size_t)(lane & 15) * ld + (lane >> 4) * 8;
  v8h lo = *(const v8h*)p;
  v8h hi = *(const v8h*)(p + 16);
  v16h r;
#pragma unroll
  for (int i = 0; i < 8; ++i) { r[i] = lo[i]; r[8 + i] = hi[i]; }
  return r;
}

// ---------------- prep: f32 [K,N] -> f16 transposed [N,K] -----------------
__global__ void transpose_to_f16(const float* __restrict__ src,
                                 _Float16* __restrict__ dst, int K, int N) {
  size_t idx = (size_t)blockIdx.x * blockDim.x + threadIdx.x;
  if (idx >= (size_t)K * N) return;
  int k = (int)(idx / N), n = (int)(idx % N);
  dst[(size_t)n * K + k] = (_Float16)src[idx];
}

// ---------------- LayerNorm -> f16 activations ----------------------------
__global__ void ln_kernel(const float* __restrict__ x,
                          const float* __restrict__ gamma,
                          const float* __restrict__ beta,
                          _Float16* __restrict__ xn) {
  int token = blockIdx.x;  // B*S tokens
  const float4* xr = (const float4*)(x + (size_t)token * D_);
  float4 v = xr[threadIdx.x];  // 256 threads * 4 = 1024
  float s = v.x + v.y + v.z + v.w;
  float s2 = v.x * v.x + v.y * v.y + v.z * v.z + v.w * v.w;
#pragma unroll
  for (int off = 16; off >= 1; off >>= 1) {
    s += __shfl_xor(s, off, 32);
    s2 += __shfl_xor(s2, off, 32);
  }
  __shared__ float red[2][8];
  int wave = threadIdx.x >> 5, lane = threadIdx.x & 31;
  if (lane == 0) { red[0][wave] = s; red[1][wave] = s2; }
  __syncthreads();
  float tot = 0.f, tot2 = 0.f;
#pragma unroll
  for (int i = 0; i < 8; ++i) { tot += red[0][i]; tot2 += red[1][i]; }
  float mu = tot * (1.0f / D_);
  float var = tot2 * (1.0f / D_) - mu * mu;
  float rinv = rsqrtf(var + 1e-5f);
  int c0 = threadIdx.x * 4;
  float vals[4] = {v.x, v.y, v.z, v.w};
  _Float16* out = xn + (size_t)token * D_ + c0;
#pragma unroll
  for (int j = 0; j < 4; ++j)
    out[j] = (_Float16)((vals[j] - mu) * rinv * gamma[c0 + j] + beta[c0 + j]);
}

// ---------------- QKV GEMM: [4096,1024] x [1024,3072] ---------------------
// Block = 8 waves (4 row-waves x 2 col-waves) -> 64x128 macro tile.
// Each wave: 16 rows x 64 cols = 4 WMMA tiles sharing one A fragment.
// K-loop double-buffered: next iteration's 10 fragment loads are in flight
// while the current 4 WMMAs issue.
__global__ void qkv_gemm(const _Float16* __restrict__ xn,
                         const _Float16* __restrict__ wT,
                         const float* __restrict__ bias,
                         _Float16* __restrict__ qb, _Float16* __restrict__ kb,
                         _Float16* __restrict__ vTb) {
  int wave = threadIdx.x >> 5, lane = threadIdx.x & 31;
  int wm = wave >> 1, wn = wave & 1;
  int m0 = blockIdx.x * 64 + wm * 16;
  int n0 = blockIdx.y * 128 + wn * 64;
  const _Float16* arow = xn + (size_t)m0 * D_;
  v8f acc[4] = {};

  v16h a_cur = load_A_frag(arow, D_);
  v16h b_cur[4];
#pragma unroll
  for (int t = 0; t < 4; ++t)
    b_cur[t] = load_BT_frag(wT + (size_t)(n0 + t * 16) * D_, D_);

  for (int kk = 0; kk < D_; kk += 32) {
    int kn = (kk + 32) & (D_ - 1);  // wraps to 0 on last iter (dead loads)
    v16h a_nxt = load_A_frag(arow + kn, D_);
    v16h b_nxt[4];
#pragma unroll
    for (int t = 0; t < 4; ++t)
      b_nxt[t] = load_BT_frag(wT + (size_t)(n0 + t * 16) * D_ + kn, D_);
#pragma unroll
    for (int t = 0; t < 4; ++t) acc[t] = wmma16x16x32(a_cur, b_cur[t], acc[t]);
    a_cur = a_nxt;
#pragma unroll
    for (int t = 0; t < 4; ++t) b_cur[t] = b_nxt[t];
  }

  int g = lane >> 4, col = lane & 15;
#pragma unroll
  for (int t = 0; t < 4; ++t) {
    int n = n0 + t * 16 + col;
    int which = n >> 10, d = n & 1023, h = d >> 6, hd = d & 63;
    float bz = bias[n];
#pragma unroll
    for (int i = 0; i < 8; ++i) {
      int token = m0 + g * 8 + i;
      int bb = token >> 11, s = token & (S_ - 1);
      _Float16 hv = (_Float16)(acc[t][i] + bz);
      size_t bh = (size_t)(bb * H_ + h);
      if (which == 0)      qb[(bh * S_ + s) * HD_ + hd] = hv;
      else if (which == 1) kb[(bh * S_ + s) * HD_ + hd] = hv;
      else                 vTb[(bh * HD_ + hd) * S_ + s] = hv;
    }
  }
}

// ---------------- Flash-style attention -----------------------------------
// Block = 8 waves, each wave owns 16 queries of one (b,h). 32 keys per step.
// Scheduling: V frags load at iteration top (consumed after softmax), next
// K-tile frags load right after the score WMMAs (double buffer), and
// global_prefetch_b8 warms K/V two tiles ahead. P is restaged through a
// per-wave-private LDS region (in-wave DS ordering; no workgroup barrier).
__global__ void attn_kernel(const _Float16* __restrict__ q,
                            const _Float16* __restrict__ k,
                            const _Float16* __restrict__ vT,
                            _Float16* __restrict__ ctx) {
  __shared__ _Float16 Pst[8][16 * 32];
  int wave = threadIdx.x >> 5, lane = threadIdx.x & 31;
  int bh = blockIdx.y;
  int q0 = blockIdx.x * 128 + wave * 16;
  const _Float16* qp = q + ((size_t)bh * S_ + q0) * HD_;
  const _Float16* kp = k + (size_t)bh * S_ * HD_;
  const _Float16* vp = vT + (size_t)bh * HD_ * S_;
  v16h aq0 = load_A_frag(qp, HD_);       // hd 0..31
  v16h aq1 = load_A_frag(qp + 32, HD_);  // hd 32..63
  v8f acc[4] = {};
  float mrow[8], lrow[8];
#pragma unroll
  for (int i = 0; i < 8; ++i) { mrow[i] = -3.0e38f; lrow[i] = 0.f; }
  const float scale = 0.125f;  // 1/sqrt(64)
  int g = lane >> 4, col = lane & 15;
  _Float16* pw = &Pst[wave][0];

  // Preload K fragments for tile 0: [sub-tile of 16 keys][hd half].
  v16h kf[4];
  kf[0] = load_BT_frag(kp, HD_);
  kf[1] = load_BT_frag(kp + 32, HD_);
  kf[2] = load_BT_frag(kp + 16 * HD_, HD_);
  kf[3] = load_BT_frag(kp + 16 * HD_ + 32, HD_);

  for (int kt = 0; kt < S_; kt += 32) {
    // V fragments for this tile — issued early, consumed after softmax.
    v16h vf[4];
#pragma unroll
    for (int t = 0; t < 4; ++t)
      vf[t] = load_BT_frag(vp + (size_t)(t * 16) * S_ + kt, S_);

    // Prefetch K/V two tiles ahead (one 128B row per lane / per instruction).
    int kpf = (kt + 64) & (S_ - 1);
    __builtin_prefetch(kp + (size_t)(kpf + lane) * HD_, 0, 3);
    __builtin_prefetch(vp + (size_t)(lane * 2) * S_ + kpf, 0, 3);
    __builtin_prefetch(vp + (size_t)(lane * 2 + 1) * S_ + kpf, 0, 3);

    v8f s0 = {}, s1 = {};
    s0 = wmma16x16x32(aq0, kf[0], s0);  // keys kt..+15
    s0 = wmma16x16x32(aq1, kf[1], s0);
    s1 = wmma16x16x32(aq0, kf[2], s1);  // keys kt+16..+31
    s1 = wmma16x16x32(aq1, kf[3], s1);

    // Double buffer: start next K-tile fragment loads now.
    int ktn = (kt + 32) & (S_ - 1);
    const _Float16* kn0 = kp + (size_t)ktn * HD_;
    kf[0] = load_BT_frag(kn0, HD_);
    kf[1] = load_BT_frag(kn0 + 32, HD_);
    kf[2] = load_BT_frag(kn0 + 16 * HD_, HD_);
    kf[3] = load_BT_frag(kn0 + 16 * HD_ + 32, HD_);

    // Online softmax; row m = g*8+i lives in the 16-lane half `g`.
#pragma unroll
    for (int i = 0; i < 8; ++i) {
      float a0 = s0[i] * scale, a1 = s1[i] * scale;
      float mx = fmaxf(a0, a1);
#pragma unroll
      for (int off = 8; off >= 1; off >>= 1)
        mx = fmaxf(mx, __shfl_xor(mx, off, 32));
      float mnew = fmaxf(mrow[i], mx);
      float corr = __expf(mrow[i] - mnew);
      float p0 = __expf(a0 - mnew);
      float p1 = __expf(a1 - mnew);
      float rs = p0 + p1;
#pragma unroll
      for (int off = 8; off >= 1; off >>= 1) rs += __shfl_xor(rs, off, 32);
      lrow[i] = lrow[i] * corr + rs;
      mrow[i] = mnew;
#pragma unroll
      for (int t = 0; t < 4; ++t) acc[t][i] *= corr;
      pw[(g * 8 + i) * 32 + col]      = (_Float16)p0;
      pw[(g * 8 + i) * 32 + 16 + col] = (_Float16)p1;
    }
    __builtin_amdgcn_wave_barrier();   // fence compiler motion around LDS
    v16h pa = load_A_frag(pw, 32);     // P as 16x32 A fragment
    __builtin_amdgcn_wave_barrier();
#pragma unroll
    for (int t = 0; t < 4; ++t) acc[t] = wmma16x16x32(pa, vf[t], acc[t]);
  }

  int b = bh >> 4, h = bh & 15;
#pragma unroll
  for (int t = 0; t < 4; ++t) {
#pragma unroll
    for (int i = 0; i < 8; ++i) {
      int token = b * S_ + q0 + g * 8 + i;
      ctx[(size_t)token * D_ + h * HD_ + t * 16 + col] =
          (_Float16)(acc[t][i] / lrow[i]);
    }
  }
}

// ---------------- out-proj + gate GEMMs fused, sigmoid + residual ---------
__global__ void outgate_gemm(const _Float16* __restrict__ ctx,
                             const _Float16* __restrict__ xn,
                             const _Float16* __restrict__ woutT,
                             const _Float16* __restrict__ wgateT,
                             const float* __restrict__ bout,
                             const float* __restrict__ bgate,
                             const float* __restrict__ x,
                             float* __restrict__ y) {
  int wave = threadIdx.x >> 5, lane = threadIdx.x & 31;
  int wm = wave >> 1, wn = wave & 1;
  int m0 = blockIdx.x * 64 + wm * 16;
  int n0 = blockIdx.y * 128 + wn * 64;
  const _Float16* crow = ctx + (size_t)m0 * D_;
  const _Float16* xrow = xn + (size_t)m0 * D_;
  v8f ao[4] = {}, ag[4] = {};

  v16h ac_cur = load_A_frag(crow, D_);
  v16h ax_cur = load_A_frag(xrow, D_);
  v16h bo_cur[4], bg_cur[4];
#pragma unroll
  for (int t = 0; t < 4; ++t) {
    bo_cur[t] = load_BT_frag(woutT + (size_t)(n0 + t * 16) * D_, D_);
    bg_cur[t] = load_BT_frag(wgateT + (size_t)(n0 + t * 16) * D_, D_);
  }

  for (int kk = 0; kk < D_; kk += 32) {
    int kn = (kk + 32) & (D_ - 1);
    v16h ac_nxt = load_A_frag(crow + kn, D_);
    v16h ax_nxt = load_A_frag(xrow + kn, D_);
    v16h bo_nxt[4], bg_nxt[4];
#pragma unroll
    for (int t = 0; t < 4; ++t) {
      bo_nxt[t] = load_BT_frag(woutT + (size_t)(n0 + t * 16) * D_ + kn, D_);
      bg_nxt[t] = load_BT_frag(wgateT + (size_t)(n0 + t * 16) * D_ + kn, D_);
    }
#pragma unroll
    for (int t = 0; t < 4; ++t) {
      ao[t] = wmma16x16x32(ac_cur, bo_cur[t], ao[t]);
      ag[t] = wmma16x16x32(ax_cur, bg_cur[t], ag[t]);
    }
    ac_cur = ac_nxt;
    ax_cur = ax_nxt;
#pragma unroll
    for (int t = 0; t < 4; ++t) { bo_cur[t] = bo_nxt[t]; bg_cur[t] = bg_nxt[t]; }
  }

  int g = lane >> 4, col = lane & 15;
#pragma unroll
  for (int t = 0; t < 4; ++t) {
    int n = n0 + t * 16 + col;
    float bo = bout[n], bg = bgate[n];
#pragma unroll
    for (int i = 0; i < 8; ++i) {
      int token = m0 + g * 8 + i;
      float o = ao[t][i] + bo;
      float z = ag[t][i] + bg;
      float gate = 1.0f / (1.0f + __expf(-z));
      y[(size_t)token * D_ + n] = o * gate + x[(size_t)token * D_ + n];
    }
  }
}

extern "C" void kernel_launch(void* const* d_in, const int* in_sizes, int n_in,
                              void* d_out, int out_size, void* d_ws,
                              size_t ws_size, hipStream_t stream) {
  const float* x     = (const float*)d_in[0];
  const float* gamma = (const float*)d_in[1];
  const float* beta  = (const float*)d_in[2];
  const float* wqkv  = (const float*)d_in[3];
  const float* bqkv  = (const float*)d_in[4];
  const float* wout  = (const float*)d_in[5];
  const float* bout  = (const float*)d_in[6];
  const float* wgate = (const float*)d_in[7];
  const float* bgate = (const float*)d_in[8];
  float* y = (float*)d_out;

  // Workspace layout (f16), total ~50 MB.
  char* ws = (char*)d_ws;
  size_t off = 0;
  _Float16* xn     = (_Float16*)(ws + off); off += (size_t)B_ * S_ * D_ * 2;
  _Float16* wqkvT  = (_Float16*)(ws + off); off += (size_t)D_ * 3 * D_ * 2;
  _Float16* woutT  = (_Float16*)(ws + off); off += (size_t)D_ * D_ * 2;
  _Float16* wgateT = (_Float16*)(ws + off); off += (size_t)D_ * D_ * 2;
  _Float16* qb     = (_Float16*)(ws + off); off += (size_t)B_ * H_ * S_ * HD_ * 2;
  _Float16* kb     = (_Float16*)(ws + off); off += (size_t)B_ * H_ * S_ * HD_ * 2;
  _Float16* vTb    = (_Float16*)(ws + off); off += (size_t)B_ * H_ * HD_ * S_ * 2;
  _Float16* ctx    = (_Float16*)(ws + off); off += (size_t)B_ * S_ * D_ * 2;

  transpose_to_f16<<<(D_ * 3 * D_ + 255) / 256, 256, 0, stream>>>(wqkv, wqkvT,
                                                                  D_, 3 * D_);
  transpose_to_f16<<<(D_ * D_ + 255) / 256, 256, 0, stream>>>(wout, woutT, D_,
                                                              D_);
  transpose_to_f16<<<(D_ * D_ + 255) / 256, 256, 0, stream>>>(wgate, wgateT,
                                                              D_, D_);
  ln_kernel<<<B_ * S_, 256, 0, stream>>>(x, gamma, beta, xn);
  qkv_gemm<<<dim3(B_ * S_ / 64, 3 * D_ / 128), 256, 0, stream>>>(
      xn, wqkvT, bqkv, qb, kb, vTb);
  attn_kernel<<<dim3(S_ / 128, B_ * H_), 256, 0, stream>>>(qb, kb, vTb, ctx);
  outgate_gemm<<<dim3(B_ * S_ / 64, D_ / 128), 256, 0, stream>>>(
      ctx, xn, woutT, wgateT, bout, bgate, x, y);
}
